// GearNet_Edge_30889404793313
// MI455X (gfx1250) — compile-verified, matrix operand order checked by reading, not run on hardware
//
#include <hip/hip_runtime.h>
#include <hip/hip_bf16.h>

// ---------------------------------------------------------------------------
// GearNet-Edge forward for MI455X (gfx1250, wave32, WMMA).
//
// All matmuls are bf16 WMMA GEMMs (v_wmma_f32_16x16x32_bf16, f32 accum) with a
// double-buffered LDS pipeline: 128x128 block tile, K-step 32, global->LDS
// staging (ds_store_b128) overlapped with compute, fragments via ds_load_b128.
// Pipeline is peeled: (nk-1) steady-state iterations with unconditional
// prefetch + store + split-barrier, then one compute-only tail iteration.
// Relation-typed aggregation is transform-then-scatter:
//   out[dst] += (X @ W[r])[src]  for edges of type r   (global_atomic_add_f32)
// identical FLOPs to the reference einsum, no (E*R,512) aggregation buffer.
// ---------------------------------------------------------------------------

typedef __attribute__((ext_vector_type(16))) __bf16 v16bf;
typedef __attribute__((ext_vector_type(8)))  __bf16 v8bf;
typedef __attribute__((ext_vector_type(8)))  float  v8f;

#define GN_N   16384
#define GN_E   131072
#define GN_E2  262144
#define GN_G   64
#define GN_RN  7
#define GN_RE  8
#define GN_D   512

#define GEMM_BIAS 1
#define GEMM_ACC  2
#define GEMM_RELU 4

#define LDS_LD 40   // 32 K-elements padded to 40 (80B => 20-bank stride, conflict-min)

static __device__ __forceinline__ float atomAddF(float* p, float v) {
  return __hip_atomic_fetch_add(p, v, __ATOMIC_RELAXED, __HIP_MEMORY_SCOPE_AGENT);
}

// Build a 16x32 bf16 fragment from two 16B loads at q and q+16 (A layout; B
// uses the same layout on Bt rows). Lane L<16: row M=L, K={0..7,16..23};
// lane L>=16: row M=L-16, K={8..15,24..31}   (CDNA5 ISA 7.12.2).
static __device__ __forceinline__ v16bf frag16(const __bf16* __restrict__ q) {
  v8bf lo = *(const v8bf*)(q);
  v8bf hi = *(const v8bf*)(q + 16);
  return __builtin_shufflevector(lo, hi, 0,1,2,3,4,5,6,7,8,9,10,11,12,13,14,15);
}

// C(M x Ncols) f32 = A(M x K) bf16 [row-major, lda] @ Bt(Ncols x K) bf16 [row-major, ldb]
// Block tile 128x128 (8 waves, each 32x64 = 2x4 WMMA tiles), K-step 32,
// double-buffered LDS staging. Requires: M mult of 128 (buffers padded),
// Bt rows padded to grid, K mult of 32, lda/ldb mult of 8.
__launch_bounds__(256)
__global__ void gemm_bf16_wmma(const __bf16* __restrict__ A, int lda,
                               const __bf16* __restrict__ Bt, int ldb,
                               float* __restrict__ C, int ldc,
                               int M, int Ncols, int K,
                               const float* __restrict__ bias, int flags) {
  __shared__ alignas(16) __bf16 sA[2][128 * LDS_LD];
  __shared__ alignas(16) __bf16 sB[2][128 * LDS_LD];

  const int tid  = threadIdx.x;
  const int lane = tid & 31;
  const int wave = tid >> 5;
  const int wm = wave >> 1;                 // 0..3
  const int wn = wave & 1;                  // 0..1
  const int mBase = blockIdx.y * 128;
  const int nBase = blockIdx.x * 128;

  // Staging: 128 rows x 64B per tile = 512 x 16B chunks; 256 threads take 2 each.
  const int r0 = tid >> 2,         p0 = tid & 3;          // rows 0..63
  const int r1 = (tid + 256) >> 2, p1 = tid & 3;          // rows 64..127
  const __bf16* gA0 = A  + (size_t)(mBase + r0) * lda + p0 * 8;
  const __bf16* gA1 = A  + (size_t)(mBase + r1) * lda + p1 * 8;
  const __bf16* gB0 = Bt + (size_t)(nBase + r0) * ldb + p0 * 8;
  const __bf16* gB1 = Bt + (size_t)(nBase + r1) * ldb + p1 * 8;
  const int l0 = r0 * LDS_LD + p0 * 8;
  const int l1 = r1 * LDS_LD + p1 * 8;

  v8f acc[2][4] = {};
  const int nk = K >> 5;

  const int frow = lane & 15;
  const int kk   = (lane >> 4) << 3;
  const int ar = wm * 32 + frow;
  const int br = wn * 64 + frow;

  // One K-step of compute from LDS buffer `cur` (12 ds_load_b128 + 8 WMMA).
  auto compute = [&](int cur) {
    const __bf16* pa = &sA[cur][0];
    const __bf16* pb = &sB[cur][0];
    v16bf fA0 = frag16(pa + (size_t)(ar +  0) * LDS_LD + kk);
    v16bf fA1 = frag16(pa + (size_t)(ar + 16) * LDS_LD + kk);
    v16bf fB0 = frag16(pb + (size_t)(br +  0) * LDS_LD + kk);
    v16bf fB1 = frag16(pb + (size_t)(br + 16) * LDS_LD + kk);
    v16bf fB2 = frag16(pb + (size_t)(br + 32) * LDS_LD + kk);
    v16bf fB3 = frag16(pb + (size_t)(br + 48) * LDS_LD + kk);
    acc[0][0] = __builtin_amdgcn_wmma_f32_16x16x32_bf16(false, fA0, false, fB0, (short)0, acc[0][0], false, false);
    acc[0][1] = __builtin_amdgcn_wmma_f32_16x16x32_bf16(false, fA0, false, fB1, (short)0, acc[0][1], false, false);
    acc[0][2] = __builtin_amdgcn_wmma_f32_16x16x32_bf16(false, fA0, false, fB2, (short)0, acc[0][2], false, false);
    acc[0][3] = __builtin_amdgcn_wmma_f32_16x16x32_bf16(false, fA0, false, fB3, (short)0, acc[0][3], false, false);
    acc[1][0] = __builtin_amdgcn_wmma_f32_16x16x32_bf16(false, fA1, false, fB0, (short)0, acc[1][0], false, false);
    acc[1][1] = __builtin_amdgcn_wmma_f32_16x16x32_bf16(false, fA1, false, fB1, (short)0, acc[1][1], false, false);
    acc[1][2] = __builtin_amdgcn_wmma_f32_16x16x32_bf16(false, fA1, false, fB2, (short)0, acc[1][2], false, false);
    acc[1][3] = __builtin_amdgcn_wmma_f32_16x16x32_bf16(false, fA1, false, fB3, (short)0, acc[1][3], false, false);
  };

  // Preload tile 0.
  {
    v8bf a0 = *(const v8bf*)gA0, a1 = *(const v8bf*)gA1;
    v8bf b0 = *(const v8bf*)gB0, b1 = *(const v8bf*)gB1;
    *(v8bf*)&sA[0][l0] = a0;  *(v8bf*)&sA[0][l1] = a1;
    *(v8bf*)&sB[0][l0] = b0;  *(v8bf*)&sB[0][l1] = b1;
  }
  __syncthreads();

  // Steady state: unconditional prefetch of tile kt+1 overlapped with compute
  // on tile kt; barrier of iteration kt-1 guarantees buf[nxt] is free.
  int kt = 0;
  for (; kt < nk - 1; ++kt) {
    const int cur = kt & 1, nxt = cur ^ 1;
    const int koff = (kt + 1) << 5;
    v8bf a0 = *(const v8bf*)(gA0 + koff), a1 = *(const v8bf*)(gA1 + koff);
    v8bf b0 = *(const v8bf*)(gB0 + koff), b1 = *(const v8bf*)(gB1 + koff);
    compute(cur);
    *(v8bf*)&sA[nxt][l0] = a0;  *(v8bf*)&sA[nxt][l1] = a1;
    *(v8bf*)&sB[nxt][l0] = b0;  *(v8bf*)&sB[nxt][l1] = b1;
    __syncthreads();
  }
  // Tail: compute-only.
  compute(kt & 1);

  // C/D layout: VGPR t holds M = t (lanes 0-15) / t+8 (lanes 16-31); N = lane&15.
  const int colBase = lane & 15;
  const int rowOff  = (lane >> 4) * 8;
  const int m0 = mBase + wm * 32;
  const int n0 = nBase + wn * 64;
  #pragma unroll
  for (int i = 0; i < 2; i++) {
    #pragma unroll
    for (int j = 0; j < 4; j++) {
      const int col = n0 + j * 16 + colBase;
      if (col >= Ncols) continue;
      const float bval = (flags & GEMM_BIAS) ? bias[col] : 0.f;
      v8f c = acc[i][j];
      #pragma unroll
      for (int t = 0; t < 8; t++) {
        const int row = m0 + i * 16 + rowOff + t;
        if (row < M) {
          const size_t idx = (size_t)row * ldc + col;
          float v = c[t] + bval;
          if (flags & GEMM_ACC)  v += C[idx];
          if (flags & GEMM_RELU) v = fmaxf(v, 0.f);
          C[idx] = v;
        }
      }
    }
  }
}

// out[dst[e]] += rows[src[e] or e]  (512-wide f32 rows), optional relation filter.
// 128 threads/block; each thread: one b128 load + 4 global_atomic_add_f32.
__global__ void scatter_add_f32(const int* __restrict__ srcIdx,
                                const int* __restrict__ dstIdx,
                                const int* __restrict__ rel, int r,
                                const float* __restrict__ rows,
                                float* __restrict__ out) {
  const int e = blockIdx.x;
  if (rel && rel[e] != r) return;
  const int s = srcIdx ? srcIdx[e] : e;
  const int d = dstIdx[e];
  const float4* rp = (const float4*)(rows + (size_t)s * GN_D);
  float* op = out + (size_t)d * GN_D;
  const int c = threadIdx.x;           // 0..127
  float4 v = rp[c];
  atomAddF(op + c * 4 + 0, v.x);
  atomAddF(op + c * 4 + 1, v.y);
  atomAddF(op + c * 4 + 2, v.z);
  atomAddF(op + c * 4 + 3, v.w);
}

// ReLU in place (f32) + bf16 mirror copy.
__global__ void relu_convert(float* __restrict__ X, __bf16* __restrict__ Y, size_t n) {
  size_t i = (size_t)blockIdx.x * blockDim.x + threadIdx.x;
  if (i < n) { float v = fmaxf(X[i], 0.f); X[i] = v; Y[i] = (__bf16)v; }
}

// dst[m*Kpad + k] = k<K ? bf16(src[m*srcld + k]) : 0      blockDim.x == Kpad
__global__ void pad_convert_rows(const float* __restrict__ src, int srcld, int K,
                                 __bf16* __restrict__ dst, int Kpad) {
  const int m = blockIdx.x, k = threadIdx.x;
  float v = (k < K) ? src[(size_t)m * srcld + k] : 0.f;
  dst[(size_t)m * Kpad + k] = (__bf16)v;
}

// dst[n*Kpad + k] = (k<K && n<Nc) ? bf16(src[k*Nc + n]) : 0   blockDim.x == Kpad
__global__ void transpose_pad(const float* __restrict__ src, int K, int Nc,
                              __bf16* __restrict__ dst, int Kpad) {
  const int n = blockIdx.x, k = threadIdx.x;
  float v = 0.f;
  if (k < K && n < Nc) v = src[(size_t)k * Nc + n];
  dst[(size_t)n * Kpad + k] = (__bf16)v;
}

// Column sums / sum-of-squares over 64-row chunks (block = 512 threads = columns).
__global__ void bn_stats(const float* __restrict__ X, float* __restrict__ sum,
                         float* __restrict__ sumsq) {
  const int col = threadIdx.x;
  const int row0 = blockIdx.x * 64;
  float s = 0.f, q = 0.f;
  for (int i = 0; i < 64; i++) {
    float v = X[(size_t)(row0 + i) * GN_D + col];
    s += v; q += v * v;
  }
  atomAddF(&sum[col], s);
  atomAddF(&sumsq[col], q);
}

__global__ void bn_apply(const float* __restrict__ X, float invM,
                         const float* __restrict__ sum, const float* __restrict__ sumsq,
                         const float* __restrict__ g, const float* __restrict__ b,
                         int do_relu, float* __restrict__ Yf, __bf16* __restrict__ Ybf) {
  const int col = threadIdx.x, row = blockIdx.x;
  const float mu  = sum[col] * invM;
  const float var = sumsq[col] * invM - mu * mu;
  float v = (X[(size_t)row * GN_D + col] - mu) * rsqrtf(var + 1e-5f) * g[col] + b[col];
  if (do_relu) v = fmaxf(v, 0.f);
  Yf [(size_t)row * GN_D + col] = v;
  Ybf[(size_t)row * GN_D + col] = (__bf16)v;
}

__global__ void pool_count(const int* __restrict__ batch, float* __restrict__ counts) {
  const int i = blockIdx.x * blockDim.x + threadIdx.x;
  if (i < GN_N) atomAddF(&counts[batch[i]], 1.0f);
}

__global__ void pool_accum(const int* __restrict__ batch, const float* __restrict__ hf,
                           float* __restrict__ pooled) {
  const int n = blockIdx.x;
  const int g = batch[n];
  const float4* rp = (const float4*)(hf + (size_t)n * GN_D);
  float* op = pooled + (size_t)g * GN_D;
  const int c = threadIdx.x;           // 0..127
  float4 v = rp[c];
  atomAddF(op + c * 4 + 0, v.x);
  atomAddF(op + c * 4 + 1, v.y);
  atomAddF(op + c * 4 + 2, v.z);
  atomAddF(op + c * 4 + 3, v.w);
}

__global__ void pool_finish(const float* __restrict__ pooled, const float* __restrict__ counts,
                            __bf16* __restrict__ Abf) {
  const int g = blockIdx.x, c = threadIdx.x;
  Abf[(size_t)g * GN_D + c] = (__bf16)(pooled[(size_t)g * GN_D + c] / counts[g]);
}

__global__ void copy_out_300(const float* __restrict__ z, float* __restrict__ out) {
  const int g = blockIdx.x, c = threadIdx.x;
  if (c < 300) out[g * 300 + c] = z[g * 320 + c];
}

// ---------------------------------------------------------------------------

extern "C" void kernel_launch(void* const* d_in, const int* in_sizes, int n_in,
                              void* d_out, int out_size, void* d_ws, size_t ws_size,
                              hipStream_t stream) {
  (void)in_sizes; (void)n_in; (void)out_size; (void)ws_size;

  auto F = [&](int i) { return (const float*)d_in[i]; };
  auto I = [&](int i) { return (const int*)d_in[i]; };

  // d_in map (top-level dict in insertion order; params flattened as a jax pytree:
  // nested dict keys sorted, lists in order):
  //  0:x  1:edge_attr
  //  2..7: bn1.b bn1.g bn2.b bn2.g bn3.b bn3.g
  //  8+4i: convs[i].W  9+4i: .We  10+4i: .Wr  11+4i: .b   (i=0..5)
  //  32+3i: emps[i].W  33+3i: .Wr  34+3i: .b              (i=0..5)
  //  50..55: head.W1 W2 W3 b1 b2 b3
  //  56: edge_index(2,E)  57: edge_type  58: edge_message_index(2,E2)
  //  59: edge_message_relation  60: batch
  const float* x_in  = F(0);
  const float* ea_in = F(1);
  const float* bnb[3] = { F(2), F(4), F(6) };
  const float* bng[3] = { F(3), F(5), F(7) };
  const int bnsel[6] = { 0, 1, 2, 2, 2, 1 };
  const float *convW[6], *convWe[6], *convWr[6], *convB[6];
  const float *empW[6], *empWr[6], *empB[6];
  for (int i = 0; i < 6; i++) {
    convW[i] = F(8 + 4 * i); convWe[i] = F(9 + 4 * i);
    convWr[i] = F(10 + 4 * i); convB[i] = F(11 + 4 * i);
    empW[i] = F(32 + 3 * i); empWr[i] = F(33 + 3 * i); empB[i] = F(34 + 3 * i);
  }
  const float *hW1 = F(50), *hW2 = F(51), *hW3 = F(52);
  const float *hb1 = F(53), *hb2 = F(54), *hb3 = F(55);
  const int* esrc  = I(56);           const int* edst  = I(56) + GN_E;
  const int* etype = I(57);
  const int* emsrc = I(58);           const int* emdst = I(58) + GN_E2;
  const int* emrel = I(59);
  const int* batch = I(60);

  // K padding per layer (mult of 32)
  const int convK[6]  = { 22, 512, 512, 512, 512, 512 };
  const int convKp[6] = { 32, 512, 512, 512, 512, 512 };
  const int empK[6]   = { 53, 512, 512, 512, 512, 512 };
  const int empKp[6]  = { 64, 512, 512, 512, 512, 512 };

  // ---- workspace carve ----
  char* base = (char*)d_ws;
  size_t off = 0;
  auto carve = [&](size_t bytes) -> void* {
    void* p = base + off;
    off = (off + bytes + 255) & ~(size_t)255;
    return p;
  };
  __bf16* e_bf    = (__bf16*)carve((size_t)GN_E * GN_D * 2);
  __bf16* h_bf    = (__bf16*)carve((size_t)GN_N * GN_D * 2);
  float*  h_f     = (float*) carve((size_t)GN_N * GN_D * 4);
  __bf16* A0x     = (__bf16*)carve((size_t)GN_N * 32 * 2);
  __bf16* A0e     = (__bf16*)carve((size_t)GN_E * 64 * 2);
  float*  tmp     = (float*) carve((size_t)GN_E * GN_D * 4);
  float*  e_out   = (float*) carve((size_t)GN_E * GN_D * 4);
  float*  msg     = (float*) carve((size_t)GN_N * GN_D * 4);
  float*  esum    = (float*) carve((size_t)GN_N * GN_D * 4);
  __bf16* esum_bf = (__bf16*)carve((size_t)GN_N * GN_D * 2);
  float*  stats   = (float*) carve(1024 * 4);
  float*  pooled  = (float*) carve((size_t)GN_G * GN_D * 4);
  float*  counts  = (float*) carve(GN_G * 4);
  __bf16* A_head  = (__bf16*)carve((size_t)128 * GN_D * 2);
  float*  z1      = (float*) carve((size_t)128 * GN_D * 4);
  __bf16* z1bf    = (__bf16*)carve((size_t)128 * GN_D * 2);
  float*  z2      = (float*) carve((size_t)128 * 320 * 4);
  __bf16* z2bf    = (__bf16*)carve((size_t)128 * 320 * 2);
  float*  z3      = (float*) carve((size_t)128 * 320 * 4);
  __bf16 *cWt[6], *cWrt[6], *cWet[6], *eWt[6], *eWrt[6];
  for (int i = 0; i < 6; i++) {
    cWt[i]  = (__bf16*)carve((size_t)GN_RN * GN_D * convKp[i] * 2);
    cWrt[i] = (__bf16*)carve((size_t)GN_D * convKp[i] * 2);
    cWet[i] = (__bf16*)carve((size_t)GN_D * GN_D * 2);
    eWt[i]  = (__bf16*)carve((size_t)GN_RE * GN_D * empKp[i] * 2);
    eWrt[i] = (__bf16*)carve((size_t)GN_D * empKp[i] * 2);
  }
  __bf16* W1t = (__bf16*)carve((size_t)512 * 512 * 2);
  __bf16* W2t = (__bf16*)carve((size_t)384 * 512 * 2);   // 300 cols padded to 384 rows
  __bf16* W3t = (__bf16*)carve((size_t)384 * 320 * 2);

  auto gemm = [&](const __bf16* A, int lda, const __bf16* Bt, int ldb,
                  float* C, int ldc, int M, int Nc, int K,
                  const float* bias, int flags) {
    dim3 g((Nc + 127) / 128, (M + 127) / 128);
    gemm_bf16_wmma<<<g, 256, 0, stream>>>(A, lda, Bt, ldb, C, ldc, M, Nc, K, bias, flags);
  };

  // ---- weight / input prep (bf16, transposed, K-padded) ----
  for (int i = 0; i < 6; i++) {
    for (int r = 0; r < GN_RN; r++)
      transpose_pad<<<GN_D, convKp[i], 0, stream>>>(
          convW[i] + (size_t)r * convK[i] * GN_D, convK[i], GN_D,
          cWt[i] + (size_t)r * GN_D * convKp[i], convKp[i]);
    transpose_pad<<<GN_D, convKp[i], 0, stream>>>(convWr[i], convK[i], GN_D, cWrt[i], convKp[i]);
    transpose_pad<<<GN_D, GN_D, 0, stream>>>(convWe[i], GN_D, GN_D, cWet[i], GN_D);
    for (int r = 0; r < GN_RE; r++)
      transpose_pad<<<GN_D, empKp[i], 0, stream>>>(
          empW[i] + (size_t)r * empK[i] * GN_D, empK[i], GN_D,
          eWt[i] + (size_t)r * GN_D * empKp[i], empKp[i]);
    transpose_pad<<<GN_D, empKp[i], 0, stream>>>(empWr[i], empK[i], GN_D, eWrt[i], empKp[i]);
  }
  transpose_pad<<<512, 512, 0, stream>>>(hW1, 512, 512, W1t, 512);
  transpose_pad<<<384, 512, 0, stream>>>(hW2, 512, 300, W2t, 512);
  transpose_pad<<<384, 320, 0, stream>>>(hW3, 300, 300, W3t, 320);
  pad_convert_rows<<<GN_N, 32, 0, stream>>>(x_in, 22, 22, A0x, 32);
  pad_convert_rows<<<GN_E, 64, 0, stream>>>(ea_in, 53, 53, A0e, 64);

  // ---- 6 layers ----
  for (int i = 0; i < 6; i++) {
    const __bf16* Ae = (i == 0) ? A0e : e_bf;  const int ldaE = empKp[i];
    const __bf16* Ah = (i == 0) ? A0x : h_bf;  const int ldaN = convKp[i];

    // edge message passing: e_out = Ae@Wr + b ; += scatter_r(Ae@W[r])
    gemm(Ae, ldaE, eWrt[i], ldaE, e_out, GN_D, GN_E, GN_D, ldaE, empB[i], GEMM_BIAS);
    for (int r = 0; r < GN_RE; r++) {
      gemm(Ae, ldaE, eWt[i] + (size_t)r * GN_D * ldaE, ldaE, tmp, GN_D, GN_E, GN_D, ldaE, nullptr, 0);
      scatter_add_f32<<<GN_E2, 128, 0, stream>>>(emsrc, emdst, emrel, r, tmp, e_out);
    }
    relu_convert<<<((size_t)GN_E * GN_D + 255) / 256, 256, 0, stream>>>(e_out, e_bf, (size_t)GN_E * GN_D);

    // node conv: msg = Ah@Wr + b ; += scatter_r(Ah@W[r])
    gemm(Ah, ldaN, cWrt[i], ldaN, msg, GN_D, GN_N, GN_D, ldaN, convB[i], GEMM_BIAS);
    for (int r = 0; r < GN_RN; r++) {
      gemm(Ah, ldaN, cWt[i] + (size_t)r * GN_D * ldaN, ldaN, tmp, GN_D, GN_N, GN_D, ldaN, nullptr, 0);
      scatter_add_f32<<<GN_E, 128, 0, stream>>>(esrc, edst, etype, r, tmp, msg);
    }

    // edge contribution: esum[n] = sum_{e: dst=n} relu(e) ; msg += esum @ We
    hipMemsetAsync(esum, 0, (size_t)GN_N * GN_D * 4, stream);
    scatter_add_f32<<<GN_E, 128, 0, stream>>>(nullptr, edst, nullptr, 0, e_out, esum);
    pad_convert_rows<<<GN_N, GN_D, 0, stream>>>(esum, GN_D, GN_D, esum_bf, GN_D);
    gemm(esum_bf, GN_D, cWet[i], GN_D, msg, GN_D, GN_N, GN_D, GN_D, nullptr, GEMM_ACC);

    // batchnorm (+ relu for i<5) -> h_f / h_bf
    hipMemsetAsync(stats, 0, 1024 * 4, stream);
    bn_stats<<<GN_N / 64, GN_D, 0, stream>>>(msg, stats, stats + 512);
    bn_apply<<<GN_N, GN_D, 0, stream>>>(msg, 1.0f / GN_N, stats, stats + 512,
                                        bng[bnsel[i]], bnb[bnsel[i]],
                                        (i < 5) ? 1 : 0, h_f, h_bf);
  }

  // ---- global mean pool + head ----
  hipMemsetAsync(counts, 0, GN_G * 4, stream);
  hipMemsetAsync(pooled, 0, (size_t)GN_G * GN_D * 4, stream);
  pool_count<<<(GN_N + 255) / 256, 256, 0, stream>>>(batch, counts);
  pool_accum<<<GN_N, 128, 0, stream>>>(batch, h_f, pooled);
  hipMemsetAsync(A_head, 0, (size_t)128 * GN_D * 2, stream);   // rows 64..127 stay zero
  pool_finish<<<GN_G, GN_D, 0, stream>>>(pooled, counts, A_head);

  gemm(A_head, GN_D, W1t, GN_D, z1, GN_D, 128, GN_D, GN_D, hb1, GEMM_BIAS | GEMM_RELU);
  pad_convert_rows<<<128, GN_D, 0, stream>>>(z1, GN_D, GN_D, z1bf, GN_D);
  gemm(z1bf, GN_D, W2t, GN_D, z2, 320, 128, 300, GN_D, hb2, GEMM_BIAS);
  pad_convert_rows<<<128, 320, 0, stream>>>(z2, 320, 300, z2bf, 320);
  gemm(z2bf, 320, W3t, 320, z3, 320, 128, 300, 320, hb3, GEMM_BIAS);
  copy_out_300<<<GN_G, 320, 0, stream>>>(z3, (float*)d_out);
}
